// TopicDecoder_62852551410382
// MI455X (gfx1250) — matrix operand order repaired and checked by reading, not compiled, not run on hardware
//
#include <hip/hip_runtime.h>
#include <hip/hip_bf16.h>

#define V 50257
#define H 1024
#define S 1024
#define B 32
#define T 256
#define TV 1000

typedef __attribute__((ext_vector_type(16))) __bf16 v16bf;
typedef __attribute__((ext_vector_type(8)))  float  v8f;

// ---------------------------------------------------------------------------
// Fragment loaders for V_WMMA_F32_16X16X32_BF16 (wave32).
// A 16x32 (MxK): lane = M (0..15, both halves), per-lane K run:
//   elements 0..15 -> K = kb + half*16 .. +15      (contiguous)
// B 32x16 (KxN): lane = N, per-lane K run:
//   elements 0..15 -> K = kb + half*16 .. +15      (contiguous)
// ---------------------------------------------------------------------------
__device__ __forceinline__ void load_frags(const float* __restrict__ Arow,
                                           const float* __restrict__ Wrow,
                                           int kb, int half,
                                           v16bf& af, v16bf& bfr)
{
    const float4* ap0 = (const float4*)(Arow + kb + half * 16);
    const float4* ap1 = (const float4*)(Arow + kb + half * 16 + 8);
    float4 a0 = ap0[0], a1 = ap0[1];
    float4 a2 = ap1[0], a3 = ap1[1];
    const float4* wp = (const float4*)(Wrow + kb + half * 16);
    float4 w0 = wp[0], w1 = wp[1], w2 = wp[2], w3 = wp[3];

    float aa[16] = {a0.x, a0.y, a0.z, a0.w, a1.x, a1.y, a1.z, a1.w,
                    a2.x, a2.y, a2.z, a2.w, a3.x, a3.y, a3.z, a3.w};
    float ww[16] = {w0.x, w0.y, w0.z, w0.w, w1.x, w1.y, w1.z, w1.w,
                    w2.x, w2.y, w2.z, w2.w, w3.x, w3.y, w3.z, w3.w};
#pragma unroll
    for (int e = 0; e < 16; ++e) {
        af[e]  = (__bf16)aa[e];
        bfr[e] = (__bf16)ww[e];
    }
}

// Tail loader: address-clamped loads + value select (no divergent branches).
__device__ __forceinline__ void load_frags_tail(const float* __restrict__ Arow,
                                                const float* __restrict__ Wrow,
                                                int kb, int half, int K,
                                                v16bf& af, v16bf& bfr)
{
#pragma unroll
    for (int j = 0; j < 8; ++j) {
        int ka0 = kb + half * 16 + j * 2;
        int ka1 = ka0 + 1;
        int kn0 = kb + half * 16 + j * 2;
        int kn1 = kn0 + 1;
        float va0 = Arow[ka0 < K ? ka0 : K - 1];
        float va1 = Arow[ka1 < K ? ka1 : K - 1];
        float vw0 = Wrow[kn0 < K ? kn0 : K - 1];
        float vw1 = Wrow[kn1 < K ? kn1 : K - 1];
        af[2 * j]      = (__bf16)(ka0 < K ? va0 : 0.0f);
        af[2 * j + 1]  = (__bf16)(ka1 < K ? va1 : 0.0f);
        bfr[2 * j]     = (__bf16)(kn0 < K ? vw0 : 0.0f);
        bfr[2 * j + 1] = (__bf16)(kn1 < K ? vw1 : 0.0f);
    }
}

// ---------------------------------------------------------------------------
// Generic skinny GEMM:  C[m,n] = act( sum_k A[m,k] * W[n, wcol0+k] + bias[n] )
// M must be a multiple of 16. One wave per (mt,nt) 16x16 tile; mt fastest so
// waves in a block share the W-fragment stream. Dual accumulator chains so
// back-to-back v_wmma ops are independent.
// ---------------------------------------------------------------------------
__global__ __launch_bounds__(256)
void k_wmma_gemm(const float* __restrict__ A, int lda, int M,
                 const float* __restrict__ W, long long ldw, long long wcol0, int N,
                 int K, const float* __restrict__ bias, int act,
                 float* __restrict__ C, long long ldc,
                 int numMT, int numNT)
{
    int wave  = blockIdx.x * (blockDim.x >> 5) + (threadIdx.x >> 5);
    int total = numMT * numNT;
    if (wave >= total) return;                 // uniform per wave: EXEC stays full
    int mt = wave % numMT;
    int nt = wave / numMT;

    int lane = threadIdx.x & 31;
    int lrow = lane & 15;
    int half = lane >> 4;

    int n  = nt * 16 + lrow;
    int nc = (n < N) ? n : (N - 1);            // clamp for loads, guard stores

    const float* __restrict__ Arow = A + (size_t)(mt * 16 + lrow) * lda;
    const float* __restrict__ Wrow = W + (size_t)nc * ldw + wcol0;

    v8f acc0 = {}, acc1 = {};
    v16bf af0, bf0, af1, bf1;

    int K64 = K & ~63;
    int K32 = K & ~31;
    int kb = 0;
    for (; kb < K64; kb += 64) {
        __builtin_prefetch(Wrow + kb + 256, 0, 3);
        load_frags(Arow, Wrow, kb,      half, af0, bf0);
        load_frags(Arow, Wrow, kb + 32, half, af1, bf1);
        acc0 = __builtin_amdgcn_wmma_f32_16x16x32_bf16(
                   false, af0, false, bf0, (short)0, acc0, false, false);
        acc1 = __builtin_amdgcn_wmma_f32_16x16x32_bf16(
                   false, af1, false, bf1, (short)0, acc1, false, false);
    }
    if (kb < K32) {                            // one full unguarded 32-block
        load_frags(Arow, Wrow, kb, half, af0, bf0);
        acc0 = __builtin_amdgcn_wmma_f32_16x16x32_bf16(
                   false, af0, false, bf0, (short)0, acc0, false, false);
        kb += 32;
    }
    if (kb < K) {                              // clamped tail block
        load_frags_tail(Arow, Wrow, kb, half, K, af1, bf1);
        acc1 = __builtin_amdgcn_wmma_f32_16x16x32_bf16(
                   false, af1, false, bf1, (short)0, acc1, false, false);
    }

    float bval = (bias != nullptr && n < N) ? bias[n] : 0.0f;
#pragma unroll
    for (int r = 0; r < 8; ++r) {
        int m = mt * 16 + r + 8 * half;        // C/D layout: VGPR r -> M = r + 8*half
        if (n < N) {
            float o = acc0[r] + acc1[r] + bval;
            if (act) o = tanhf(o);
            C[(size_t)m * ldc + n] = o;
        }
    }
}

// ---------------------------------------------------------------------------
// x[b,h] = embedding[input_step[b], h]
// ---------------------------------------------------------------------------
__global__ void k_gather(const int* __restrict__ idx,
                         const float* __restrict__ emb,
                         float* __restrict__ x)
{
    int i = blockIdx.x * 256 + threadIdx.x;    // B*H
    int b = i >> 10, h = i & 1023;
    x[i] = emb[(size_t)idx[b] * H + h];
}

// ---------------------------------------------------------------------------
// GRU gate combine; writes h_new to ws, to d_out hidden slot, and concat[:,0:H)
// ---------------------------------------------------------------------------
__global__ void k_gru_combine(const float* __restrict__ gx,
                              const float* __restrict__ gh,
                              const float* __restrict__ h_prev,
                              float* __restrict__ h_new,
                              float* __restrict__ out_hidden,
                              float* __restrict__ concat)
{
    int i = blockIdx.x * 256 + threadIdx.x;    // B*H
    int b = i >> 10, h = i & 1023;
    size_t g = (size_t)b * 3 * H + h;
    float r = 1.0f / (1.0f + expf(-(gx[g] + gh[g])));
    float z = 1.0f / (1.0f + expf(-(gx[g + H] + gh[g + H])));
    float n = tanhf(gx[g + 2 * H] + r * gh[g + 2 * H]);
    float hn = (1.0f - z) * n + z * h_prev[i];
    h_new[i] = hn;
    out_hidden[i] = hn;
    concat[(size_t)b * 4048 + h] = hn;
}

// ---------------------------------------------------------------------------
// energies[b,s] = dot(h_new[b,:], enc[s,b,:])   (first 128MB encoder pass)
// ---------------------------------------------------------------------------
__global__ void k_energies(const float* __restrict__ h_new,
                           const float* __restrict__ enc,
                           float* __restrict__ energ)
{
    int b = blockIdx.y;
    __shared__ float4 sh[H / 4];
    const float4* hb = (const float4*)(h_new + (size_t)b * H);
    sh[threadIdx.x] = hb[threadIdx.x];
    __syncthreads();
    int s = blockIdx.x * 256 + threadIdx.x;
    const float4* row = (const float4*)(enc + ((size_t)s * B + b) * H);
    float acc = 0.0f;
    for (int k = 0; k < H / 4; ++k) {
        float4 a = sh[k], e = row[k];
        acc += a.x * e.x + a.y * e.y + a.z * e.z + a.w * e.w;
    }
    energ[(size_t)b * S + s] = acc;
}

// ---------------------------------------------------------------------------
// In-place row softmax: one block per row.
// ---------------------------------------------------------------------------
__global__ void k_softmax(float* __restrict__ data, int n)
{
    float* row = data + (size_t)blockIdx.x * n;
    __shared__ float red[256];
    int tid = threadIdx.x;

    float m = -3.4e38f;
    for (int i = tid; i < n; i += 256) m = fmaxf(m, row[i]);
    red[tid] = m; __syncthreads();
    for (int o = 128; o > 0; o >>= 1) {
        if (tid < o) red[tid] = fmaxf(red[tid], red[tid + o]);
        __syncthreads();
    }
    float mx = red[0]; __syncthreads();

    float s = 0.0f;
    for (int i = tid; i < n; i += 256) {
        float e = expf(row[i] - mx);
        row[i] = e; s += e;
    }
    red[tid] = s; __syncthreads();
    for (int o = 128; o > 0; o >>= 1) {
        if (tid < o) red[tid] += red[tid + o];
        __syncthreads();
    }
    float inv = 1.0f / red[0];
    for (int i = tid; i < n; i += 256) row[i] *= inv;
}

// ---------------------------------------------------------------------------
// context[b,h] = sum_s attn[b,s]*enc[s,b,h] -> concat[:, H:2H)
// Coalesced over h; second encoder pass, should hit L2.
// ---------------------------------------------------------------------------
__global__ void k_context(const float* __restrict__ attn,
                          const float* __restrict__ enc,
                          float* __restrict__ concat)
{
    int b = blockIdx.y;
    __shared__ float w[S];
    for (int i = threadIdx.x; i < S; i += 256) w[i] = attn[(size_t)b * S + i];
    __syncthreads();
    int h = blockIdx.x * 256 + threadIdx.x;
    float acc = 0.0f;
    for (int s = 0; s < S; ++s)
        acc += w[s] * enc[((size_t)s * B + b) * H + h];
    concat[(size_t)b * 4048 + H + h] = acc;
}

// ---------------------------------------------------------------------------
// base[b,i] = h_prev[b,:]@W_h[i,:] + enc_last[b,:]@W_e[i,:] + tattn_b[i]
// (W_h = tattn_W[:, 0:H], W_e = tattn_W[:, H+TV : H+TV+H])
// ---------------------------------------------------------------------------
__global__ void k_topic_base(const float* __restrict__ h_prev,
                             const float* __restrict__ enc,
                             const float* __restrict__ tattn_W,
                             const float* __restrict__ tattn_b,
                             float* __restrict__ base)
{
    int b = blockIdx.y;
    __shared__ float hh[H];
    __shared__ float ee[H];
    for (int k = threadIdx.x; k < H; k += 256) {
        hh[k] = h_prev[(size_t)b * H + k];
        ee[k] = enc[((size_t)(S - 1) * B + b) * H + k];
    }
    __syncthreads();
    int i = blockIdx.x * 256 + threadIdx.x;
    const float* wr = tattn_W + (size_t)i * (H + TV + H);
    float acc = tattn_b[i];
    for (int k = 0; k < H; ++k)
        acc += hh[k] * wr[k] + ee[k] * wr[H + TV + k];
    base[(size_t)b * H + i] = acc;
}

// ---------------------------------------------------------------------------
// t_scores[b,t] = sum_h tanh(base[b,h] + tproj[t,h]) * v[h]
// ---------------------------------------------------------------------------
__global__ void k_tscores(const float* __restrict__ base,
                          const float* __restrict__ tproj,
                          const float* __restrict__ vvec,
                          float* __restrict__ tsc)
{
    int b = blockIdx.x;
    int t = threadIdx.x;                        // T == 256
    __shared__ float bb[H];
    __shared__ float vv[H];
    for (int k = threadIdx.x; k < H; k += 256) {
        bb[k] = base[(size_t)b * H + k];
        vv[k] = vvec[k];
    }
    __syncthreads();
    const float* tp = tproj + (size_t)t * H;
    float acc = 0.0f;
    for (int h = 0; h < H; ++h)
        acc += tanhf(bb[h] + tp[h]) * vv[h];
    tsc[(size_t)b * T + t] = acc;
}

// ---------------------------------------------------------------------------
// topic_context / Pk_context -> concat[:, 2H:2H+TV) and [:, 2H+TV:2H+2TV)
// ---------------------------------------------------------------------------
__global__ void k_topic_ctx(const float* __restrict__ tattn,
                            const float* __restrict__ codes,
                            const float* __restrict__ topics,
                            float* __restrict__ concat)
{
    int b = blockIdx.y;
    __shared__ float w[T];
    __shared__ float c[T];
    w[threadIdx.x] = tattn[(size_t)b * T + threadIdx.x];
    c[threadIdx.x] = codes[(size_t)b * T + threadIdx.x];
    __syncthreads();
    int vv = blockIdx.x * 256 + threadIdx.x;
    if (vv < TV) {
        float tc = 0.0f, pk = 0.0f;
        for (int t = 0; t < T; ++t) {
            float tval = topics[(size_t)t * TV + vv];
            tc += w[t] * tval;
            pk += c[t] * tval;
        }
        concat[(size_t)b * 4048 + 2 * H + vv]      = tc;
        concat[(size_t)b * 4048 + 2 * H + TV + vv] = pk;
    }
}

// ---------------------------------------------------------------------------

static inline int wblocks(int waves) { return (waves + 7) / 8; }

extern "C" void kernel_launch(void* const* d_in, const int* in_sizes, int n_in,
                              void* d_out, int out_size, void* d_ws, size_t ws_size,
                              hipStream_t stream) {
    const int*   input_step = (const int*)  d_in[0];
    const float* last_hidden= (const float*)d_in[1];   // (B,H)
    const float* enc        = (const float*)d_in[2];   // (S,B,H)
    const float* codes      = (const float*)d_in[3];   // (B,T)
    const float* embedding  = (const float*)d_in[4];
    const float* W_ih       = (const float*)d_in[5];
    const float* W_hh       = (const float*)d_in[6];
    const float* b_ih       = (const float*)d_in[7];
    const float* b_hh       = (const float*)d_in[8];
    const float* topics     = (const float*)d_in[9];
    const float* tattn_W    = (const float*)d_in[10];
    const float* tattn_b    = (const float*)d_in[11];
    const float* vvec       = (const float*)d_in[12];
    const float* concat_W   = (const float*)d_in[13];
    const float* concat_b   = (const float*)d_in[14];
    const float* out_W      = (const float*)d_in[15];
    const float* out_b      = (const float*)d_in[16];

    float* out_logits = (float*)d_out;                 // (B,V)
    float* out_hidden = (float*)d_out + (size_t)B * V; // (B,H)

    float* ws        = (float*)d_ws;
    float* x         = ws;               // B*H
    float* gx        = x       + B * H;  // B*3H
    float* gh        = gx      + B * 3 * H;
    float* h_new     = gh      + B * 3 * H;
    float* energ     = h_new   + B * H;  // B*S (attn weights in place)
    float* base      = energ   + B * S;  // B*H
    float* tproj     = base    + B * H;  // T*H
    float* tsc       = tproj   + T * H;  // B*T (topic attn in place)
    float* concat_in = tsc     + B * T;  // B*4048
    float* concat_o  = concat_in + B * 4048; // B*H

    // 1) embedding gather
    k_gather<<<B * H / 256, 256, 0, stream>>>(input_step, embedding, x);

    // 2) GRU gates (WMMA): gx = x@W_ih^T + b_ih ; gh = h_prev@W_hh^T + b_hh
    k_wmma_gemm<<<wblocks(2 * 192), 256, 0, stream>>>(
        x, H, B, W_ih, H, 0, 3 * H, H, b_ih, 0, gx, 3 * H, 2, 192);
    k_wmma_gemm<<<wblocks(2 * 192), 256, 0, stream>>>(
        last_hidden, H, B, W_hh, H, 0, 3 * H, H, b_hh, 0, gh, 3 * H, 2, 192);

    // 3) gate combine -> h_new (ws, out_hidden, concat[:,0:H))
    k_gru_combine<<<B * H / 256, 256, 0, stream>>>(
        gx, gh, last_hidden, h_new, out_hidden, concat_in);

    // 4) attention energies + softmax + context (two encoder passes, L2 reuse)
    k_energies<<<dim3(S / 256, B), 256, 0, stream>>>(h_new, enc, energ);
    k_softmax<<<B, 256, 0, stream>>>(energ, S);
    k_context<<<dim3(H / 256, B), 256, 0, stream>>>(energ, enc, concat_in);

    // 5) topic attention
    k_topic_base<<<dim3(H / 256, B), 256, 0, stream>>>(
        last_hidden, enc, tattn_W, tattn_b, base);
    k_wmma_gemm<<<wblocks(16 * 64), 256, 0, stream>>>(
        topics, TV, T, tattn_W, H + TV + H, H, H, TV, nullptr, 0,
        tproj, H, 16, 64);
    k_tscores<<<B, 256, 0, stream>>>(base, tproj, vvec, tsc);
    k_softmax<<<B, 256, 0, stream>>>(tsc, T);
    k_topic_ctx<<<dim3((TV + 255) / 256, B), 256, 0, stream>>>(
        tsc, codes, topics, concat_in);

    // 6) concat_output = tanh(concat_in @ concat_W^T + concat_b)
    k_wmma_gemm<<<wblocks(2 * 64), 256, 0, stream>>>(
        concat_in, 4048, B, concat_W, 4048, 0, H, 4048, concat_b, 1,
        concat_o, H, 2, 64);

    // 7) logits = concat_o @ out_W^T + out_b  (dominant: 206MB stream of out_W)
    const int numNT = (V + 15) / 16;  // 3142
    k_wmma_gemm<<<wblocks(2 * numNT), 256, 0, stream>>>(
        concat_o, H, B, out_W, H, 0, V, H, out_b, 0,
        out_logits, V, 2, numNT);

    // 8) softmax over vocab (in place in d_out)
    k_softmax<<<B, 256, 0, stream>>>(out_logits, V);
}
